// mGRU_49649821942438
// MI455X (gfx1250) — matrix operand order repaired and checked by reading, not compiled, not run on hardware
//
#include <hip/hip_runtime.h>
#include <stdint.h>

// ---------------- problem dims ----------------
#define BQ    16            // batch
#define TSEQ  512           // time steps
#define INPD  2048          // input dim
#define HID   1024          // hidden / FEAT
#define GATE  (3*HID)       // 3072
#define TWOH  (2*HID)       // 2048
#define OUTD  1024
#define MROWS (TSEQ*BQ)     // 8192 flattened (t,b) rows

// ---------------- types ----------------
typedef __attribute__((ext_vector_type(16))) __bf16 v16bf;
typedef __attribute__((ext_vector_type(8)))  __bf16 v8bf;
typedef __attribute__((ext_vector_type(8)))  float  v8f;

__device__ __forceinline__ unsigned short f32_to_bf16_bits(float f) {
    unsigned int u = __builtin_bit_cast(unsigned int, f);
    unsigned int r = u + 0x7FFFu + ((u >> 16) & 1u);   // round-to-nearest-even
    return (unsigned short)(r >> 16);
}

// =====================================================================
// Generic batched WMMA GEMM:  C[b] = A[b] (MxK, bf16) * Bt[b] (KxN, bf16) + bias[b]
// Bt is the weight ALREADY transposed to [K,N] so B-fragments load contiguous.
// mode 0: C f32 row-major [M,N]
// mode 1: C bf16 row-major [M,N]           (feeds next GEMM as A operand)
// mode 2: C f32 scattered to [BQ, N, TSEQ] with row = t*BQ + b (final output)
// Wave tile: 16(M) x 64(N); per K-step: 1 A-frag, 4 B-frags, 4 WMMAs.
// =====================================================================
__global__ void wmma_gemm_kernel(const unsigned short* __restrict__ A,  long long sA,
                                 const unsigned short* __restrict__ Bt, long long sB,
                                 const float* __restrict__ bias,        long long sBias,
                                 void* __restrict__ Cout,               long long sC,
                                 int M, int N, int K, int mode)
{
    const int batch = blockIdx.y;
    const __bf16* Ah = reinterpret_cast<const __bf16*>(A)  + (size_t)batch * sA;
    const __bf16* Bh = reinterpret_cast<const __bf16*>(Bt) + (size_t)batch * sB;
    const float*  bp = bias ? (bias + (size_t)batch * sBias) : nullptr;

    const int lane    = threadIdx.x & 31;
    const int ntiles4 = N >> 6;                       // N / 64
    const int wid     = blockIdx.x * (blockDim.x >> 5) + (threadIdx.x >> 5);
    const int mt      = wid / ntiles4;
    const int ng      = wid - mt * ntiles4;
    if (mt * 16 >= M) return;
    const int m0 = mt * 16;
    const int n0 = ng * 64;

    // ISA fragment lane mappings (wave32)
    const int arow = lane & 15;
    const int akb  = (lane < 16) ? 0 : 8;             // A: K sub-block base
    const int bkr  = (lane & 15) + ((lane < 16) ? 0 : 16); // B: K row for this lane

    v8f acc[4] = {v8f{}, v8f{}, v8f{}, v8f{}};

    for (int k0 = 0; k0 < K; k0 += 32) {
        // ---- A fragment: row arow, halves {kb..kb+7, kb+16..kb+23} ----
        const __bf16* ap = Ah + (size_t)(m0 + arow) * K + (k0 + akb);
        v8bf alo = *reinterpret_cast<const v8bf*>(ap);
        v8bf ahi = *reinterpret_cast<const v8bf*>(ap + 16);
        v16bf afrag = __builtin_shufflevector(alo, ahi,
            0,1,2,3,4,5,6,7, 8,9,10,11,12,13,14,15);

        // ---- 4 B fragments: col lane&15, K rows {kb..kb+7, kb+16..kb+23} ----
        const __bf16* bpk = Bh + (size_t)(k0 + akb) * N + n0 + (lane & 15);
        v16bf bfrag[4];
        #pragma unroll
        for (int u = 0; u < 4; ++u) {
            #pragma unroll
            for (int s = 0; s < 16; ++s)
                bfrag[u][s] = bpk[(size_t)((s & 7) + ((s >> 3) << 4)) * N + u * 16];
        }

        #pragma unroll
        for (int u = 0; u < 4; ++u)
            acc[u] = __builtin_amdgcn_wmma_f32_16x16x32_bf16(
                false, afrag, false, bfrag[u], (short)0, acc[u], false, false);
    }

    // ---- epilogue: VGPR v -> row v + (lane>=16 ? 8 : 0), col lane&15 ----
    const int rbase = (lane >> 4) << 3;
    const int ccol  = lane & 15;
    #pragma unroll
    for (int u = 0; u < 4; ++u) {
        const int ncol = n0 + u * 16 + ccol;
        const float bv = bp ? bp[ncol] : 0.0f;
        #pragma unroll
        for (int v = 0; v < 8; ++v) {
            const int mrow = m0 + rbase + v;
            const float val = acc[u][v] + bv;
            if (mode == 0) {
                ((float*)Cout)[(size_t)batch * sC + (size_t)mrow * N + ncol] = val;
            } else if (mode == 1) {
                ((unsigned short*)Cout)[(size_t)batch * sC + (size_t)mrow * N + ncol] =
                    f32_to_bf16_bits(val);
            } else {
                const int t = mrow >> 4;          // row = t*BQ + b, BQ = 16
                const int b = mrow & 15;
                ((float*)Cout)[(size_t)b * OUTD * TSEQ + (size_t)ncol * TSEQ + t] = val;
            }
        }
    }
}

// =====================================================================
// fp32 [nb, N, K] -> bf16 transposed [nb, K, N]
// =====================================================================
__global__ void transpose_to_bf16_kernel(const float* __restrict__ W,
                                         unsigned short* __restrict__ Wt,
                                         int N, int K, long long total)
{
    long long idx = (long long)blockIdx.x * blockDim.x + threadIdx.x;
    if (idx >= total) return;
    long long nk  = (long long)N * K;
    int  b  = (int)(idx / nk);
    long long r = idx - (long long)b * nk;
    int  n  = (int)(r / K);
    int  k  = (int)(r - (long long)n * K);
    Wt[(size_t)b * nk + (size_t)k * N + n] = f32_to_bf16_bits(W[idx]);
}

// inp [B, INPD, T] fp32 -> xin_bf [(t*B+b), INPD] bf16  (coalesced read)
__global__ void convert_input_kernel(const float* __restrict__ inp,
                                     unsigned short* __restrict__ xin)
{
    long long idx = (long long)blockIdx.x * blockDim.x + threadIdx.x;
    if (idx >= (long long)BQ * INPD * TSEQ) return;
    int b = (int)(idx / ((long long)INPD * TSEQ));
    long long r = idx - (long long)b * INPD * TSEQ;
    int i = (int)(r / TSEQ);
    int t = (int)(r - (long long)i * TSEQ);
    xin[((size_t)t * BQ + b) * INPD + i] = f32_to_bf16_bits(inp[idx]);
}

__global__ void init_h_kernel(float* __restrict__ h_f, unsigned short* __restrict__ h_bf)
{
    int idx = blockIdx.x * blockDim.x + threadIdx.x;
    if (idx >= 2 * BQ * HID) return;
    h_f[idx]  = 0.0f;
    h_bf[idx] = 0;   // 0x0000 == bf16(0)
}

// =====================================================================
// Per-step gate fusion (both directions).
// gx: [2, MROWS, GATE] f32 (includes b_ih). gh: [2, BQ, GATE] f32 (no bias).
// h_f/h_bf: [2, BQ, HID]. y_bf: [(t*BQ+b), 2H] with fwd in [0,H), bwd in [H,2H).
// =====================================================================
__global__ void gru_gate_kernel(const float* __restrict__ gx,
                                const float* __restrict__ gh,
                                const float* __restrict__ b_hh,  // [2, GATE]
                                float* __restrict__ h_f,
                                unsigned short* __restrict__ h_bf,
                                unsigned short* __restrict__ y_bf,
                                int step)
{
    int idx = blockIdx.x * blockDim.x + threadIdx.x;
    if (idx >= 2 * BQ * HID) return;
    int d   = idx / (BQ * HID);
    int rem = idx - d * (BQ * HID);
    int b   = rem / HID;
    int j   = rem - b * HID;
    int t   = d ? (TSEQ - 1 - step) : step;

    size_t gxb = ((size_t)d * MROWS + (size_t)t * BQ + b) * GATE;
    float xr = gx[gxb + j];
    float xz = gx[gxb + HID + j];
    float xn = gx[gxb + 2 * HID + j];

    size_t ghb = ((size_t)d * BQ + b) * GATE;
    const float* bh = b_hh + (size_t)d * GATE;
    float hr = gh[ghb + j]           + bh[j];
    float hz = gh[ghb + HID + j]     + bh[HID + j];
    float hn = gh[ghb + 2 * HID + j] + bh[2 * HID + j];

    float rg = 1.0f / (1.0f + __expf(-(xr + hr)));
    float zg = 1.0f / (1.0f + __expf(-(xz + hz)));
    float ng = tanhf(xn + rg * hn);

    size_t hidx = (size_t)(d * BQ + b) * HID + j;
    float hnew = (1.0f - zg) * ng + zg * h_f[hidx];
    h_f[hidx]  = hnew;
    unsigned short hb = f32_to_bf16_bits(hnew);
    h_bf[hidx] = hb;
    y_bf[((size_t)t * BQ + b) * TWOH + (size_t)d * HID + j] = hb;
}

// =====================================================================
extern "C" void kernel_launch(void* const* d_in, const int* in_sizes, int n_in,
                              void* d_out, int out_size, void* d_ws, size_t ws_size,
                              hipStream_t stream)
{
    const float* inp     = (const float*)d_in[0];
    const float* w_in    = (const float*)d_in[1];
    const float* b_in    = (const float*)d_in[2];
    const float* w_ih_l0 = (const float*)d_in[3];
    const float* w_hh_l0 = (const float*)d_in[4];
    const float* b_ih_l0 = (const float*)d_in[5];
    const float* b_hh_l0 = (const float*)d_in[6];
    const float* w_ih_l1 = (const float*)d_in[7];
    const float* w_hh_l1 = (const float*)d_in[8];
    const float* b_ih_l1 = (const float*)d_in[9];
    const float* b_hh_l1 = (const float*)d_in[10];
    const float* w_out   = (const float*)d_in[11];
    const float* b_out   = (const float*)d_in[12];

    // ---- workspace carve-up (256B aligned) ----
    char* ws = (char*)d_ws;
    size_t off = 0;
    auto take = [&](size_t bytes) -> char* {
        char* p = ws + off;
        off = (off + bytes + 255) & ~(size_t)255;
        return p;
    };
    unsigned short* xin_bf  = (unsigned short*)take((size_t)MROWS * INPD * 2);
    unsigned short* x1_bf   = (unsigned short*)take((size_t)MROWS * HID  * 2);
    unsigned short* y0_bf   = (unsigned short*)take((size_t)MROWS * TWOH * 2);
    unsigned short* y1_bf   = (unsigned short*)take((size_t)MROWS * TWOH * 2);
    float*          gx      = (float*)take((size_t)2 * MROWS * GATE * 4);
    float*          gh      = (float*)take((size_t)2 * BQ * GATE * 4);
    float*          h_f     = (float*)take((size_t)2 * BQ * HID * 4);
    unsigned short* h_bf    = (unsigned short*)take((size_t)2 * BQ * HID * 2);
    unsigned short* win_t   = (unsigned short*)take((size_t)INPD * HID  * 2);
    unsigned short* wih0_t  = (unsigned short*)take((size_t)2 * HID  * GATE * 2);
    unsigned short* whh0_t  = (unsigned short*)take((size_t)2 * HID  * GATE * 2);
    unsigned short* wih1_t  = (unsigned short*)take((size_t)2 * TWOH * GATE * 2);
    unsigned short* whh1_t  = (unsigned short*)take((size_t)2 * HID  * GATE * 2);
    unsigned short* wout_t  = (unsigned short*)take((size_t)TWOH * OUTD * 2);
    (void)ws_size; (void)in_sizes; (void)n_in; (void)out_size;

    auto cdiv = [](long long a, long long b) { return (int)((a + b - 1) / b); };

    auto xpose = [&](const float* W, unsigned short* Wt, int N, int K, int nb) {
        long long tot = (long long)nb * N * K;
        transpose_to_bf16_kernel<<<cdiv(tot, 256), 256, 0, stream>>>(W, Wt, N, K, tot);
    };
    auto gemm = [&](const unsigned short* A, long long sA,
                    const unsigned short* Bt, long long sB,
                    const float* bias, long long sBias,
                    void* C, long long sC,
                    int M, int N, int K, int mode, int nb) {
        int waves = (M / 16) * (N / 64);
        dim3 grid((unsigned)cdiv(waves, 8), (unsigned)nb, 1);
        wmma_gemm_kernel<<<grid, 256, 0, stream>>>(A, sA, Bt, sB, bias, sBias,
                                                   C, sC, M, N, K, mode);
    };

    // ---- 1. one-time conversions: weights -> bf16 [K,N], input -> bf16 (t,b)-rows ----
    xpose(w_in,    win_t,  HID,  INPD, 1);
    xpose(w_ih_l0, wih0_t, GATE, HID,  2);
    xpose(w_hh_l0, whh0_t, GATE, HID,  2);
    xpose(w_ih_l1, wih1_t, GATE, TWOH, 2);
    xpose(w_hh_l1, whh1_t, GATE, HID,  2);
    xpose(w_out,   wout_t, OUTD, TWOH, 1);
    {
        long long tot = (long long)BQ * INPD * TSEQ;
        convert_input_kernel<<<cdiv(tot, 256), 256, 0, stream>>>(inp, xin_bf);
    }

    // ---- 2. input projection: x1 = xin @ w_in^T + b_in  (bf16 out) ----
    gemm(xin_bf, 0, win_t, 0, b_in, 0, x1_bf, 0, MROWS, HID, INPD, /*mode*/1, 1);

    // ---- 3. layer 0: gx for both dirs, then 512 recurrent steps ----
    gemm(x1_bf, 0, wih0_t, (long long)HID * GATE, b_ih_l0, GATE,
         gx, (long long)MROWS * GATE, MROWS, GATE, HID, /*mode*/0, 2);
    init_h_kernel<<<cdiv(2 * BQ * HID, 256), 256, 0, stream>>>(h_f, h_bf);
    for (int s = 0; s < TSEQ; ++s) {
        gemm(h_bf, (long long)BQ * HID, whh0_t, (long long)HID * GATE,
             nullptr, 0, gh, (long long)BQ * GATE, BQ, GATE, HID, /*mode*/0, 2);
        gru_gate_kernel<<<cdiv(2 * BQ * HID, 256), 256, 0, stream>>>(
            gx, gh, b_hh_l0, h_f, h_bf, y0_bf, s);
    }

    // ---- 4. layer 1: gx (K = 2H), then 512 recurrent steps ----
    gemm(y0_bf, 0, wih1_t, (long long)TWOH * GATE, b_ih_l1, GATE,
         gx, (long long)MROWS * GATE, MROWS, GATE, TWOH, /*mode*/0, 2);
    init_h_kernel<<<cdiv(2 * BQ * HID, 256), 256, 0, stream>>>(h_f, h_bf);
    for (int s = 0; s < TSEQ; ++s) {
        gemm(h_bf, (long long)BQ * HID, whh1_t, (long long)HID * GATE,
             nullptr, 0, gh, (long long)BQ * GATE, BQ, GATE, HID, /*mode*/0, 2);
        gru_gate_kernel<<<cdiv(2 * BQ * HID, 256), 256, 0, stream>>>(
            gx, gh, b_hh_l1, h_f, h_bf, y1_bf, s);
    }

    // ---- 5. output projection, scattered to [B, OUT, T] ----
    gemm(y1_bf, 0, wout_t, 0, b_out, 0, d_out, 0, MROWS, OUTD, TWOH, /*mode*/2, 1);
}
